// GATBlock_3710851744314
// MI455X (gfx1250) — compile-verified, matrix operand order checked by reading, not run on hardware
//

#include <hip/hip_runtime.h>
#include <hip/hip_bf16.h>

// ---------------------------------------------------------------------------
// GAT block for MI455X (gfx1250): bf16 WMMA GEMMs + atomic segment softmax.
// ---------------------------------------------------------------------------

#define N_NODES 25000
#define N_EDGES 400000
#define ETOT    (N_EDGES + N_NODES)
#define D       256
#define HEADS   8
#define HEAD_DIM 32
#define D_FF    1024

typedef __attribute__((ext_vector_type(16))) __bf16 v16bf;
typedef __attribute__((ext_vector_type(8)))  float  v8f;

union FragBF { v16bf v; uint4 q[2]; };

__device__ __forceinline__ unsigned short f2bf(float f) {
    unsigned u = __float_as_uint(f);
    unsigned r = (u + 0x7FFFu + ((u >> 16) & 1u)) >> 16;   // RNE
    return (unsigned short)r;
}

// monotone float -> uint key for atomicMax over signed floats
__device__ __forceinline__ unsigned enc_f(float f) {
    unsigned u = __float_as_uint(f);
    return ((int)u >= 0) ? (u | 0x80000000u) : ~u;
}
__device__ __forceinline__ float dec_f(unsigned k) {
    return (k & 0x80000000u) ? __uint_as_float(k ^ 0x80000000u)
                             : __uint_as_float(~k);
}
#define ENC_NEG_INF 0x007FFFFFu   /* enc_f(-inf) */

__device__ __forceinline__ float gelu_tanh_f(float x) {
    const float c = 0.7978845608028654f; // sqrt(2/pi)
    float x3 = x * x * x;
    return 0.5f * x * (1.0f + tanhf(c * (x + 0.044715f * x3)));
}

// ---------------------------------------------------------------------------
// Weight convert: f32 [K x N] row-major  ->  bf16 [N x K] (N-major, transposed)
// ---------------------------------------------------------------------------
__global__ void k_conv_weight(const float* __restrict__ src,
                              unsigned short* __restrict__ dst, int K, int N) {
    int idx = blockIdx.x * blockDim.x + threadIdx.x;
    if (idx >= K * N) return;
    int k = idx / N, n = idx % N;
    dst[(size_t)n * K + k] = f2bf(src[idx]);
}

// ---------------------------------------------------------------------------
// LayerNorm over D=256, wave32 per row (8 floats/lane). Optional second input
// (residual sum before normalizing), optional f32 and/or bf16 outputs.
// ---------------------------------------------------------------------------
__global__ void k_layernorm(const float* __restrict__ A,
                            const float* __restrict__ B,
                            const float* __restrict__ g,
                            const float* __restrict__ bta,
                            float* __restrict__ outF,
                            unsigned short* __restrict__ outB, int M) {
    int row  = blockIdx.x * 8 + (threadIdx.x >> 5);
    int lane = threadIdx.x & 31;
    if (row >= M) return;
    const float* a = A + (size_t)row * D + lane * 8;
    float v[8];
#pragma unroll
    for (int j = 0; j < 8; j++) v[j] = a[j];
    if (B) {
        const float* bb = B + (size_t)row * D + lane * 8;
#pragma unroll
        for (int j = 0; j < 8; j++) v[j] += bb[j];
    }
    float s = 0.f, ss = 0.f;
#pragma unroll
    for (int j = 0; j < 8; j++) { s += v[j]; ss += v[j] * v[j]; }
#pragma unroll
    for (int m = 16; m > 0; m >>= 1) {
        s  += __shfl_xor(s,  m, 32);
        ss += __shfl_xor(ss, m, 32);
    }
    float mean = s * (1.0f / D);
    float var  = ss * (1.0f / D) - mean * mean;
    float rs   = rsqrtf(var + 1e-5f);
    int c0 = lane * 8;
#pragma unroll
    for (int j = 0; j < 8; j++) {
        float y = (v[j] - mean) * rs * g[c0 + j] + bta[c0 + j];
        if (outF) outF[(size_t)row * D + c0 + j] = y;
        if (outB) outB[(size_t)row * D + c0 + j] = f2bf(y);
    }
}

// ---------------------------------------------------------------------------
// bf16 WMMA GEMM: out[M,N] = act(A[M,K] @ Bt[N,K]^T + bias) (+resid)
// Each wave owns a 64x16 output strip (MT=4 16x16 tiles stacked in M):
// the B fragment is loaded once per K-step and reused by 4 WMMAs.
// Double-buffered K pipeline: fragments for step kk+32 are prefetched into
// distinct registers before the step-kk WMMAs issue, so the backend can use
// partial s_wait_loadcnt values instead of a full drain before every WMMA.
// Out-of-range A rows are CLAMPED to row M-1 (row i of A only feeds row i of
// C, and rows >= M are never stored) -> branch-free inner loop, EXEC all-1s.
// Fragment addressing follows CDNA5 16-bit A(16x32)/B(32x16) VGPR layouts:
// lanes 0-15: K {0..7,16..23}, lanes 16-31: K {8..15,24..31}, 2 x b128 loads.
// ---------------------------------------------------------------------------
template <int K, int MT>
__global__ void k_gemm_wmma(const unsigned short* __restrict__ A,
                            const unsigned short* __restrict__ Bt,
                            const float* __restrict__ bias,
                            const float* __restrict__ resid,
                            float* __restrict__ outF,
                            unsigned short* __restrict__ outB,
                            int M, int N, int act) {
    int lane = threadIdx.x & 31;
    int wave = threadIdx.x >> 5;
    int tm = blockIdx.x * (16 * MT);
    int tn = (blockIdx.y * 8 + wave) * 16;
    int r    = lane & 15;
    int koff = (lane & 16) ? 8 : 0;

    const unsigned short* bp = Bt + (size_t)(tn + r) * K + koff;
    const unsigned short* ap[MT];
#pragma unroll
    for (int t = 0; t < MT; t++) {
        int arow = tm + t * 16 + r;
        if (arow > M - 1) arow = M - 1;       // clamp (see note above)
        ap[t] = A + (size_t)arow * K + koff;
    }

    v8f acc[MT];
#pragma unroll
    for (int t = 0; t < MT; t++)
        acc[t] = (v8f){0.f, 0.f, 0.f, 0.f, 0.f, 0.f, 0.f, 0.f};

    // prologue: load K-step 0
    FragBF fb, fa[MT];
    fb.q[0] = *(const uint4*)(bp);
    fb.q[1] = *(const uint4*)(bp + 16);
#pragma unroll
    for (int t = 0; t < MT; t++) {
        fa[t].q[0] = *(const uint4*)(ap[t]);
        fa[t].q[1] = *(const uint4*)(ap[t] + 16);
    }

#pragma unroll
    for (int kk = 0; kk < K; kk += 32) {
        FragBF fbn, fan[MT];
        if (kk + 32 < K) {                    // prefetch next step
            fbn.q[0] = *(const uint4*)(bp + kk + 32);
            fbn.q[1] = *(const uint4*)(bp + kk + 48);
#pragma unroll
            for (int t = 0; t < MT; t++) {
                fan[t].q[0] = *(const uint4*)(ap[t] + kk + 32);
                fan[t].q[1] = *(const uint4*)(ap[t] + kk + 48);
            }
        }
#pragma unroll
        for (int t = 0; t < MT; t++)
            acc[t] = __builtin_amdgcn_wmma_f32_16x16x32_bf16(
                false, fa[t].v, false, fb.v, (short)0, acc[t], false, false);
        if (kk + 32 < K) {
            fb = fbn;
#pragma unroll
            for (int t = 0; t < MT; t++) fa[t] = fan[t];
        }
    }

    int col = tn + r;
    float bval = bias ? bias[col] : 0.f;
    int rsub = (lane & 16) ? 8 : 0;
#pragma unroll
    for (int t = 0; t < MT; t++) {
#pragma unroll
        for (int i = 0; i < 8; i++) {
            int row = tm + t * 16 + rsub + i;
            if (row >= M) continue;
            float v = acc[t][i] + bval;
            if (act) v = gelu_tanh_f(v);
            if (resid) v += resid[(size_t)row * N + col];
            if (outF) outF[(size_t)row * N + col] = v;
            if (outB) outB[(size_t)row * N + col] = f2bf(v);
        }
    }
}

// ---------------------------------------------------------------------------
// Per-(node,head) attention coefficients: a = <h[n,h,:], att[h,:]>
// ---------------------------------------------------------------------------
__global__ void k_attn_coef(const float* __restrict__ h,
                            const float* __restrict__ att_src,
                            const float* __restrict__ att_dst,
                            float* __restrict__ asrc,
                            float* __restrict__ adst) {
    int idx = blockIdx.x * blockDim.x + threadIdx.x;
    if (idx >= N_NODES * HEADS) return;
    int n = idx >> 3, hh = idx & 7;
    const float* hp = h + (size_t)n * D + hh * HEAD_DIM;
    const float* ws = att_src + hh * HEAD_DIM;
    const float* wd = att_dst + hh * HEAD_DIM;
    float ss = 0.f, sd = 0.f;
#pragma unroll
    for (int d = 0; d < HEAD_DIM; d++) {
        float v = hp[d];
        ss += v * ws[d];
        sd += v * wd[d];
    }
    asrc[idx] = ss;
    adst[idx] = sd;
}

__global__ void k_init_stats(float* __restrict__ agg,
                             float* __restrict__ denom,
                             unsigned* __restrict__ emax) {
    int idx = blockIdx.x * blockDim.x + threadIdx.x;
    if (idx < N_NODES * D) agg[idx] = 0.f;
    if (idx < N_NODES * HEADS) { denom[idx] = 0.f; emax[idx] = ENC_NEG_INF; }
}

__device__ __forceinline__ void edge_sd(const int* ei, int e, int& s, int& d) {
    if (e < N_EDGES) { s = ei[e]; d = ei[N_EDGES + e]; }
    else             { s = d = e - N_EDGES; }   // self loops
}

__global__ void k_edge_max(const int* __restrict__ ei,
                           const float* __restrict__ asrc,
                           const float* __restrict__ adst,
                           float* __restrict__ ebuf,
                           unsigned* __restrict__ emax) {
    int idx = blockIdx.x * blockDim.x + threadIdx.x;
    if (idx >= ETOT * HEADS) return;
    int e = idx >> 3, hh = idx & 7, s, d;
    edge_sd(ei, e, s, d);
    float v = asrc[s * HEADS + hh] + adst[d * HEADS + hh];
    v = (v > 0.f) ? v : 0.2f * v;                // leaky_relu
    ebuf[idx] = v;
    atomicMax(&emax[d * HEADS + hh], enc_f(v));
}

__global__ void k_edge_expsum(const int* __restrict__ ei,
                              const unsigned* __restrict__ emax,
                              float* __restrict__ ebuf,
                              float* __restrict__ denom) {
    int idx = blockIdx.x * blockDim.x + threadIdx.x;
    if (idx >= ETOT * HEADS) return;
    int e = idx >> 3, hh = idx & 7, s, d;
    edge_sd(ei, e, s, d);
    float ee = expf(ebuf[idx] - dec_f(emax[d * HEADS + hh]));
    ebuf[idx] = ee;
    atomicAdd(&denom[d * HEADS + hh], ee);
}

// wave32 per edge: lane owns 8 contiguous features (one head per 4 lanes)
__global__ void k_edge_agg(const int* __restrict__ ei,
                           const float* __restrict__ h,
                           const float* __restrict__ ebuf,
                           const float* __restrict__ denom,
                           float* __restrict__ agg) {
    int edge = blockIdx.x * 8 + (threadIdx.x >> 5);
    if (edge >= ETOT) return;
    int lane = threadIdx.x & 31;
    int s, d;
    edge_sd(ei, edge, s, d);
    int hh = lane >> 2;
    float alpha = ebuf[edge * HEADS + hh] / denom[d * HEADS + hh];
    const float* hp = h + (size_t)s * D + lane * 8;
    float*       ap = agg + (size_t)d * D + lane * 8;
    float4 h0 = *(const float4*)hp;
    float4 h1 = *(const float4*)(hp + 4);
    atomicAdd(ap + 0, alpha * h0.x);
    atomicAdd(ap + 1, alpha * h0.y);
    atomicAdd(ap + 2, alpha * h0.z);
    atomicAdd(ap + 3, alpha * h0.w);
    atomicAdd(ap + 4, alpha * h1.x);
    atomicAdd(ap + 5, alpha * h1.y);
    atomicAdd(ap + 6, alpha * h1.z);
    atomicAdd(ap + 7, alpha * h1.w);
}

// ---------------------------------------------------------------------------
extern "C" void kernel_launch(void* const* d_in, const int* in_sizes, int n_in,
                              void* d_out, int out_size, void* d_ws, size_t ws_size,
                              hipStream_t stream) {
    (void)in_sizes; (void)n_in; (void)out_size; (void)ws_size;
    const float* x        = (const float*)d_in[0];
    const int*   ei       = (const int*)  d_in[1];
    const float* W        = (const float*)d_in[2];
    const float* att_src  = (const float*)d_in[3];
    const float* att_dst  = (const float*)d_in[4];
    const float* bias     = (const float*)d_in[5];
    const float* W_res    = (const float*)d_in[6];
    const float* ln1_g    = (const float*)d_in[7];
    const float* ln1_b    = (const float*)d_in[8];
    const float* ln2_g    = (const float*)d_in[9];
    const float* ln2_b    = (const float*)d_in[10];
    const float* ffn_w1   = (const float*)d_in[11];
    const float* ffn_b1   = (const float*)d_in[12];
    const float* ffn_w2   = (const float*)d_in[13];
    const float* ffn_b2   = (const float*)d_in[14];
    float* out = (float*)d_out;

    // -------- workspace layout (bytes, 256-aligned) --------
    char* base = (char*)d_ws;
    size_t off = 0;
    auto alloc = [&](size_t bytes) {
        size_t r = off;
        off = (off + bytes + 255) & ~(size_t)255;
        return r;
    };
    size_t o_wtW   = alloc((size_t)D * D * 2);          // bf16 [N][K]
    size_t o_wtR   = alloc((size_t)D * D * 2);
    size_t o_wtF1  = alloc((size_t)D_FF * D * 2);       // [1024][256]
    size_t o_wtF2  = alloc((size_t)D * D_FF * 2);       // [256][1024]
    size_t o_xnbf  = alloc((size_t)N_NODES * D * 2);
    size_t o_asrc  = alloc((size_t)N_NODES * HEADS * 4);
    size_t o_adst  = alloc((size_t)N_NODES * HEADS * 4);
    size_t o_emax  = alloc((size_t)N_NODES * HEADS * 4);
    size_t o_den   = alloc((size_t)N_NODES * HEADS * 4);
    size_t o_h     = alloc((size_t)N_NODES * D * 4);    // |-- contiguous span
    size_t o_ebuf  = alloc((size_t)ETOT * HEADS * 4);   // |   reused later by
    size_t o_agg   = alloc((size_t)N_NODES * D * 4);    // |   ff1 (bf16, 51MB)
    size_t o_res   = alloc((size_t)N_NODES * D * 4);
    size_t o_oln   = alloc((size_t)N_NODES * D * 4);
    size_t o_olnbf = alloc((size_t)N_NODES * D * 2);
    size_t o_ff1bf = o_h;  // alias dead h+ebuf+agg region (64.8MB >= 51.2MB)

    unsigned short* wtW   = (unsigned short*)(base + o_wtW);
    unsigned short* wtR   = (unsigned short*)(base + o_wtR);
    unsigned short* wtF1  = (unsigned short*)(base + o_wtF1);
    unsigned short* wtF2  = (unsigned short*)(base + o_wtF2);
    unsigned short* xnbf  = (unsigned short*)(base + o_xnbf);
    float*    asrc  = (float*)(base + o_asrc);
    float*    adst  = (float*)(base + o_adst);
    unsigned* emax  = (unsigned*)(base + o_emax);
    float*    den   = (float*)(base + o_den);
    float*    hbuf  = (float*)(base + o_h);
    float*    ebuf  = (float*)(base + o_ebuf);
    float*    agg   = (float*)(base + o_agg);
    float*    res   = (float*)(base + o_res);
    float*    oln   = (float*)(base + o_oln);
    unsigned short* olnbf = (unsigned short*)(base + o_olnbf);
    unsigned short* ff1bf = (unsigned short*)(base + o_ff1bf);

    const int B = 256;
    // 1) convert + transpose weights to bf16 N-major
    k_conv_weight<<<(D * D + B - 1) / B, B, 0, stream>>>(W, wtW, D, D);
    k_conv_weight<<<(D * D + B - 1) / B, B, 0, stream>>>(W_res, wtR, D, D);
    k_conv_weight<<<(D * D_FF + B - 1) / B, B, 0, stream>>>(ffn_w1, wtF1, D, D_FF);
    k_conv_weight<<<(D_FF * D + B - 1) / B, B, 0, stream>>>(ffn_w2, wtF2, D_FF, D);

    // 2) LN1 -> xn (bf16)
    k_layernorm<<<(N_NODES + 7) / 8, B, 0, stream>>>(x, nullptr, ln1_g, ln1_b,
                                                     nullptr, xnbf, N_NODES);

    dim3 gemm_grid((N_NODES + 63) / 64, D / 128);       // 64-row strips
    // 3) h = xn @ W (f32 out)
    k_gemm_wmma<D, 4><<<gemm_grid, B, 0, stream>>>(xnbf, wtW, nullptr, nullptr,
                                                   hbuf, nullptr, N_NODES, D, 0);
    // 4) res = xn @ W_res + bias
    k_gemm_wmma<D, 4><<<gemm_grid, B, 0, stream>>>(xnbf, wtR, bias, nullptr,
                                                   res, nullptr, N_NODES, D, 0);
    // 5) per-(node,head) attention coefficients
    k_attn_coef<<<(N_NODES * HEADS + B - 1) / B, B, 0, stream>>>(hbuf, att_src,
                                                                 att_dst, asrc, adst);
    // 6) zero agg/denom, emax = -inf
    k_init_stats<<<(N_NODES * D + B - 1) / B, B, 0, stream>>>(agg, den, emax);
    // 7-9) segment softmax + aggregation over E+N edges
    int nEH = ETOT * HEADS;
    k_edge_max<<<(nEH + B - 1) / B, B, 0, stream>>>(ei, asrc, adst, ebuf, emax);
    k_edge_expsum<<<(nEH + B - 1) / B, B, 0, stream>>>(ei, emax, ebuf, den);
    k_edge_agg<<<(ETOT + 7) / 8, B, 0, stream>>>(ei, hbuf, ebuf, den, agg);

    // 10) out_ln = LN2(agg + res) -> f32 + bf16
    k_layernorm<<<(N_NODES + 7) / 8, B, 0, stream>>>(agg, res, ln2_g, ln2_b,
                                                     oln, olnbf, N_NODES);
    // 11) ff1 = gelu(out_ln @ ffn_w1 + b1) (bf16 out)
    dim3 gemm_grid_ff((N_NODES + 63) / 64, D_FF / 128);
    k_gemm_wmma<D, 4><<<gemm_grid_ff, B, 0, stream>>>(olnbf, wtF1, ffn_b1, nullptr,
                                                      nullptr, ff1bf, N_NODES, D_FF, 1);
    // 12) out = out_ln + ff1 @ ffn_w2 + b2
    k_gemm_wmma<D_FF, 4><<<gemm_grid, B, 0, stream>>>(ff1bf, wtF2, ffn_b2, oln,
                                                      out, nullptr, N_NODES, D, 0);
}
